// CrossAttention_71090298683697
// MI455X (gfx1250) — matrix operand order repaired and checked by reading, not compiled, not run on hardware
//
#include <hip/hip_runtime.h>

// ---------------------------------------------------------------------------
// CDNA5 (gfx1250) cross-attention: bf16 WMMA everywhere, f32 accumulate.
// Round 2: software-pipelined fragment loads (double-buffered registers) so
// s_wait_loadcnt stops draining to 0 before each WMMA; global_prefetch hints
// on the K/V streams.
// ---------------------------------------------------------------------------

typedef __attribute__((ext_vector_type(16))) __bf16 bf16x16;
typedef __attribute__((ext_vector_type(8)))  __bf16 bf16x8;
typedef __attribute__((ext_vector_type(8)))  float  f32x8;

constexpr int Bn  = 4;
constexpr int SQ  = 2048;
constexpr int SKV = 2048;
constexpr int D   = 1024;
constexpr int H   = 16;
constexpr int DH  = 64;

static __device__ __forceinline__ f32x8 wmma_bf16(bf16x16 a, bf16x16 b, f32x8 c) {
  // D = A(16x32) * B(32x16) + C(16x16 f32)
  return __builtin_amdgcn_wmma_f32_16x16x32_bf16(
      /*neg_a=*/false, a, /*neg_b=*/false, b,
      /*c_mod=*/(short)0, c, /*reuse_a=*/false, /*reuse_b=*/false);
}

// A-operand fragment: row-major A, 16x32 tile starting at p0 (tile row 0, k 0).
// ISA layout: lane l -> row (l&15); element e: k = 8*hi + (e<8 ? e : e+8).
static __device__ __forceinline__ bf16x16 ldA(const __bf16* p0, int ld, int lane) {
  int m = lane & 15, hi = lane >> 4;
  const __bf16* p = p0 + (size_t)m * ld + hi * 8;
  bf16x8 a0 = *(const bf16x8*)(p);
  bf16x8 a1 = *(const bf16x8*)(p + 16);
  bf16x16 r;
#pragma unroll
  for (int i = 0; i < 8; ++i) { r[i] = a0[i]; r[i + 8] = a1[i]; }
  return r;
}

// B-operand fragment from a TRANSPOSED B (Bt row n = output col, contraction
// contiguous). ISA layout: lane l -> col (l&15); element e: k = 8*hi + (e<8 ? e : e+8).
static __device__ __forceinline__ bf16x16 ldB(const __bf16* p0, int ld, int lane) {
  int n = lane & 15, hi = lane >> 4;
  const __bf16* p = p0 + (size_t)n * ld + hi * 8;
  bf16x8 b0 = *(const bf16x8*)(p);
  bf16x8 b1 = *(const bf16x8*)(p + 16);
  bf16x16 r;
#pragma unroll
  for (int i = 0; i < 8; ++i) { r[i] = b0[i]; r[i + 8] = b1[i]; }
  return r;
}

// --------------------------- prep kernels ----------------------------------

__global__ void k_cvt(const float* __restrict__ s, __bf16* __restrict__ d, int n) {
  int i = blockIdx.x * 256 + threadIdx.x;
  if (i < n) d[i] = (__bf16)s[i];
}

// Wt[n][k] = W[k][n]  (f32 -> bf16), D x D
__global__ void k_transpose_w(const float* __restrict__ W, __bf16* __restrict__ Wt) {
  int i = blockIdx.x * 256 + threadIdx.x;   // i < D*D
  int n = i >> 10, k = i & (D - 1);
  Wt[i] = (__bf16)W[(size_t)k * D + n];
}

// --------------------------- projection GEMM -------------------------------
// C[M=8192][N=1024] = A(bf16,row-major) * Wt^T ; epilogue scatters per mode:
//   mode 0/1 (Q,K): dst[b][h][s][dh]    (bf16)
//   mode 2   (V)  : dst[b][h][dh][s]    (bf16, transposed for P*V operand)
__global__ void __launch_bounds__(256)
k_proj(const __bf16* __restrict__ A, const __bf16* __restrict__ Wt,
       __bf16* __restrict__ dst, int mode) {
  int lane = threadIdx.x & 31;
  int wave = blockIdx.x * 8 + (threadIdx.x >> 5);   // 8192 waves
  int mt = wave >> 4, nt = wave & 15;               // 512 x 16 tiles
  int m0 = mt * 16, n0 = nt * 64;

  const __bf16* Arow = A + (size_t)m0 * D;

  f32x8 acc[4] = {};
  // pipeline stage 0
  bf16x16 af = ldA(Arow, D, lane);
  bf16x16 bfr[4];
#pragma unroll
  for (int t = 0; t < 4; ++t)
    bfr[t] = ldB(Wt + (size_t)(n0 + t * 16) * D, D, lane);

#pragma unroll 2
  for (int k = 0; k < D; k += 32) {
    int kn = (k + 32 < D) ? k + 32 : k;   // clamped prefetch index
    // issue next-step loads BEFORE consuming current fragments
    bf16x16 af_n = ldA(Arow + kn, D, lane);
    bf16x16 bn[4];
#pragma unroll
    for (int t = 0; t < 4; ++t)
      bn[t] = ldB(Wt + (size_t)(n0 + t * 16) * D + kn, D, lane);

#pragma unroll
    for (int t = 0; t < 4; ++t)
      acc[t] = wmma_bf16(af, bfr[t], acc[t]);

    af = af_n;
#pragma unroll
    for (int t = 0; t < 4; ++t) bfr[t] = bn[t];
  }

  int nn = lane & 15, hi = lane >> 4;
#pragma unroll
  for (int t = 0; t < 4; ++t) {
#pragma unroll
    for (int r = 0; r < 8; ++r) {
      int m  = m0 + r + 8 * hi;             // global row = b*S + s
      int gn = n0 + t * 16 + nn;            // global col = h*64 + dh
      int b = m >> 11, s = m & (SQ - 1);
      int h = gn >> 6, dh = gn & (DH - 1);
      size_t o;
      if (mode == 2) o = (((size_t)b * H + h) * DH + dh) * (size_t)SKV + s;
      else           o = (((size_t)b * H + h) * (size_t)SQ + s) * DH + dh;
      dst[o] = (__bf16)acc[t][r];
    }
  }
}

// --------------------------- flash attention -------------------------------
// One wave = one (b, h, 16-row q tile). Online softmax over kv blocks of 32.
__global__ void __launch_bounds__(256)
k_attn(const __bf16* __restrict__ Q, const __bf16* __restrict__ K,
       const __bf16* __restrict__ Vt, __bf16* __restrict__ O) {
  __shared__ __bf16 lds_p[8][16 * 32];      // per-wave P staging (C->A reshuffle)

  int lane  = threadIdx.x & 31;
  int wslot = threadIdx.x >> 5;
  int wave  = blockIdx.x * 8 + wslot;       // 8192 waves total
  int qt = wave & 127;                      // Sq/16 = 128
  int bh = wave >> 7;                       // b*H + h
  int h = bh & (H - 1), b = bh >> 4;

  const __bf16* Qhead = Q + ((size_t)bh * SQ + qt * 16) * DH;
  const __bf16* Khead = K + (size_t)bh * SKV * DH;
  const __bf16* Vhead = Vt + (size_t)bh * DH * SKV;

  bf16x16 aq0 = ldA(Qhead + 0,  DH, lane);  // dh 0..31
  bf16x16 aq1 = ldA(Qhead + 32, DH, lane);  // dh 32..63

  f32x8 oacc[4] = {};
  float rmax[8], rsum[8];
#pragma unroll
  for (int r = 0; r < 8; ++r) { rmax[r] = -3.0e38f; rsum[r] = 0.0f; }

  const float scale = 0.125f;               // 1/sqrt(64)
  int nn = lane & 15, hi = lane >> 4;
  __bf16* pp = lds_p[wslot];

  // preload K fragments for the first kv block
  bf16x16 kb[4];
  kb[0] = ldB(Khead + 0,                     DH, lane);
  kb[1] = ldB(Khead + 32,                    DH, lane);
  kb[2] = ldB(Khead + (size_t)16 * DH + 0,   DH, lane);
  kb[3] = ldB(Khead + (size_t)16 * DH + 32,  DH, lane);

  for (int j = 0; j < SKV; j += 32) {
    // ---- V fragments for this block (independent of softmax) --------------
    const __bf16* Vb = Vhead + j;
    bf16x16 bv[4];
#pragma unroll
    for (int t = 0; t < 4; ++t)
      bv[t] = ldB(Vb + (size_t)(t * 16) * SKV, SKV, lane);

    // ---- scores: two 16x16 tiles (kv cols j..j+15, j+16..j+31) -------------
    f32x8 s0 = {}, s1 = {};
    s0 = wmma_bf16(aq0, kb[0], s0);
    s0 = wmma_bf16(aq1, kb[1], s0);
    s1 = wmma_bf16(aq0, kb[2], s1);
    s1 = wmma_bf16(aq1, kb[3], s1);

    // ---- prefetch K fragments for next block (clamped) ---------------------
    {
      int jn = (j + 32 < SKV) ? j + 32 : j;
      const __bf16* Kn0 = Khead + (size_t)jn * DH;
      const __bf16* Kn1 = Kn0 + (size_t)16 * DH;
      kb[0] = ldB(Kn0 + 0,  DH, lane);
      kb[1] = ldB(Kn0 + 32, DH, lane);
      kb[2] = ldB(Kn1 + 0,  DH, lane);
      kb[3] = ldB(Kn1 + 32, DH, lane);
      // L2 prefetch hints two blocks ahead (emits global_prefetch_b8)
      int jp = (j + 64 < SKV) ? j + 64 : j;
      __builtin_prefetch(Khead + (size_t)(jp + lane) * DH, 0, 0);
      __builtin_prefetch(Vhead + (size_t)lane * SKV + jp, 0, 0);
    }

    // ---- online softmax (overlaps the loads above) -------------------------
    float bm[8];
#pragma unroll
    for (int r = 0; r < 8; ++r) {
      s0[r] *= scale; s1[r] *= scale;
      bm[r] = fmaxf(s0[r], s1[r]);
    }
#pragma unroll
    for (int mask = 1; mask <= 8; mask <<= 1)
#pragma unroll
      for (int r = 0; r < 8; ++r)
        bm[r] = fmaxf(bm[r], __shfl_xor(bm[r], mask, 32));

    float alpha[8];
#pragma unroll
    for (int r = 0; r < 8; ++r) {
      float mn = fmaxf(rmax[r], bm[r]);
      alpha[r] = __expf(rmax[r] - mn);
      rmax[r] = mn;
      s0[r] = __expf(s0[r] - mn);
      s1[r] = __expf(s1[r] - mn);
    }
    float bs[8];
#pragma unroll
    for (int r = 0; r < 8; ++r) bs[r] = s0[r] + s1[r];
#pragma unroll
    for (int mask = 1; mask <= 8; mask <<= 1)
#pragma unroll
      for (int r = 0; r < 8; ++r)
        bs[r] += __shfl_xor(bs[r], mask, 32);
#pragma unroll
    for (int r = 0; r < 8; ++r) {
      rsum[r] = rsum[r] * alpha[r] + bs[r];
      oacc[0][r] *= alpha[r]; oacc[1][r] *= alpha[r];
      oacc[2][r] *= alpha[r]; oacc[3][r] *= alpha[r];
    }

    // ---- reshuffle P from C-layout into A-operand layout via LDS -----------
#pragma unroll
    for (int r = 0; r < 8; ++r) {
      int m = r + 8 * hi;
      pp[m * 32 + nn]      = (__bf16)s0[r];
      pp[m * 32 + 16 + nn] = (__bf16)s1[r];
    }
    asm volatile("s_wait_dscnt 0" ::: "memory");
    bf16x16 ap;
    {
      const __bf16* q = pp + (lane & 15) * 32 + hi * 8;
#pragma unroll
      for (int i = 0; i < 8; ++i) { ap[i] = q[i]; ap[i + 8] = q[i + 16]; }
    }

    // ---- P * V (V stored transposed: rows = dh, contraction kv contiguous) -
#pragma unroll
    for (int t = 0; t < 4; ++t)
      oacc[t] = wmma_bf16(ap, bv[t], oacc[t]);
  }

  // ---- normalize + store attn output [B, Sq, H*Dh] as bf16 -----------------
  float inv[8];
#pragma unroll
  for (int r = 0; r < 8; ++r) inv[r] = 1.0f / rsum[r];
#pragma unroll
  for (int t = 0; t < 4; ++t)
#pragma unroll
    for (int r = 0; r < 8; ++r) {
      int m = qt * 16 + r + 8 * hi;
      size_t o = ((size_t)b * SQ + m) * D + h * DH + t * 16 + nn;
      O[o] = (__bf16)(oacc[t][r] * inv[r]);
    }
}

// --------------------------- output projection -----------------------------
__global__ void __launch_bounds__(256)
k_oproj(const __bf16* __restrict__ A, const __bf16* __restrict__ Wt,
        const float* __restrict__ bias, float* __restrict__ out) {
  int lane = threadIdx.x & 31;
  int wave = blockIdx.x * 8 + (threadIdx.x >> 5);
  int mt = wave >> 4, nt = wave & 15;
  int m0 = mt * 16, n0 = nt * 64;

  const __bf16* Arow = A + (size_t)m0 * D;

  f32x8 acc[4] = {};
  bf16x16 af = ldA(Arow, D, lane);
  bf16x16 bfr[4];
#pragma unroll
  for (int t = 0; t < 4; ++t)
    bfr[t] = ldB(Wt + (size_t)(n0 + t * 16) * D, D, lane);

#pragma unroll 2
  for (int k = 0; k < D; k += 32) {
    int kn = (k + 32 < D) ? k + 32 : k;
    bf16x16 af_n = ldA(Arow + kn, D, lane);
    bf16x16 bn[4];
#pragma unroll
    for (int t = 0; t < 4; ++t)
      bn[t] = ldB(Wt + (size_t)(n0 + t * 16) * D + kn, D, lane);

#pragma unroll
    for (int t = 0; t < 4; ++t)
      acc[t] = wmma_bf16(af, bfr[t], acc[t]);

    af = af_n;
#pragma unroll
    for (int t = 0; t < 4; ++t) bfr[t] = bn[t];
  }

  int nn = lane & 15, hi = lane >> 4;
#pragma unroll
  for (int t = 0; t < 4; ++t) {
#pragma unroll
    for (int r = 0; r < 8; ++r) {
      int m  = m0 + r + 8 * hi;
      int gn = n0 + t * 16 + nn;
      out[(size_t)m * D + gn] = acc[t][r] + bias[gn];
    }
  }
}

// ---------------------------------------------------------------------------

extern "C" void kernel_launch(void* const* d_in, const int* in_sizes, int n_in,
                              void* d_out, int out_size, void* d_ws, size_t ws_size,
                              hipStream_t stream) {
  (void)in_sizes; (void)n_in; (void)out_size; (void)ws_size;

  const float* inputs  = (const float*)d_in[0];  // [4,2048,1024]
  const float* context = (const float*)d_in[1];  // [4,2048,1024]
  const float* Wq = (const float*)d_in[2];
  const float* Wk = (const float*)d_in[3];
  const float* Wv = (const float*)d_in[4];
  const float* Wo = (const float*)d_in[5];
  const float* bo = (const float*)d_in[6];
  float* out = (float*)d_out;

  const size_t XSZ = (size_t)Bn * SQ * D * 2;   // 16 MiB each
  const size_t WSZ = (size_t)D * D * 2;         // 2 MiB each
  char* ws = (char*)d_ws;
  size_t off = 0;
  __bf16* Xbf  = (__bf16*)(ws + off); off += XSZ;
  __bf16* Cbf  = (__bf16*)(ws + off); off += XSZ;
  __bf16* Wqt  = (__bf16*)(ws + off); off += WSZ;
  __bf16* Wkt  = (__bf16*)(ws + off); off += WSZ;
  __bf16* Wvt  = (__bf16*)(ws + off); off += WSZ;
  __bf16* Wot  = (__bf16*)(ws + off); off += WSZ;
  __bf16* Qb   = (__bf16*)(ws + off); off += XSZ;
  __bf16* Kb   = (__bf16*)(ws + off); off += XSZ;
  __bf16* Vtb  = (__bf16*)(ws + off); off += XSZ;
  __bf16* AOb  = (__bf16*)(ws + off); off += XSZ;

  const int NX = Bn * SQ * D;                   // 8388608
  k_cvt<<<NX / 256, 256, 0, stream>>>(inputs,  Xbf, NX);
  k_cvt<<<NX / 256, 256, 0, stream>>>(context, Cbf, NX);
  k_transpose_w<<<(D * D) / 256, 256, 0, stream>>>(Wq, Wqt);
  k_transpose_w<<<(D * D) / 256, 256, 0, stream>>>(Wk, Wkt);
  k_transpose_w<<<(D * D) / 256, 256, 0, stream>>>(Wv, Wvt);
  k_transpose_w<<<(D * D) / 256, 256, 0, stream>>>(Wo, Wot);

  // 8192 waves each: (8192/16 m-tiles) x (1024/64 n-tiles), 8 waves/block
  k_proj<<<1024, 256, 0, stream>>>(Xbf, Wqt, Qb,  0);
  k_proj<<<1024, 256, 0, stream>>>(Cbf, Wkt, Kb,  1);
  k_proj<<<1024, 256, 0, stream>>>(Cbf, Wvt, Vtb, 2);

  // 8192 waves: B*H*(Sq/16) q-tiles
  k_attn<<<1024, 256, 0, stream>>>(Qb, Kb, Vtb, AOb);

  k_oproj<<<1024, 256, 0, stream>>>(AOb, Wot, bo, out);
}
